// AGRU_48584670052560
// MI455X (gfx1250) — matrix operand order repaired and checked by reading, not compiled, not run on hardware
//
#include <hip/hip_runtime.h>

#define S_LEN 512
#define B_DIM 64
#define I_DIM 512
#define H_DIM 1024
#define GAMMA_C 0.01f
#define RSTR (H_DIM + 8)   // padded LDS column stride (halfs) for recurrence

typedef __attribute__((ext_vector_type(16))) __bf16 v16bf;
typedef __attribute__((ext_vector_type(8)))  float  v8f;

struct alignas(16) U128 { unsigned x, y, z, w; };
union Frag { v16bf v; U128 q[2]; };

__device__ __forceinline__ unsigned short f2bf(float f) {
  unsigned u = __float_as_uint(f);
  u += 0x7FFFu + ((u >> 16) & 1u);   // round-to-nearest-even
  return (unsigned short)(u >> 16);
}
__device__ __forceinline__ float sigm(float x) { return 1.0f / (1.0f + __expf(-x)); }

__device__ __forceinline__ v8f zero8() {
  v8f z;
#pragma unroll
  for (int i = 0; i < 8; ++i) z[i] = 0.0f;
  return z;
}

// A fragment: 16x32 bf16, lane l -> row m0+(l&15); elems 0..7 = K[k0+8*hl ..],
// elems 8..15 = K[k0+16+8*hl ..]  (ISA 7.12.2, 16-bit A 16x32)
__device__ __forceinline__ v16bf load_a_frag(const unsigned short* base, int row, int K, int k0, int hl) {
  Frag a;
  const unsigned short* p = base + (size_t)row * K + k0 + 8 * hl;
  a.q[0] = *reinterpret_cast<const U128*>(p);
  a.q[1] = *reinterpret_cast<const U128*>(p + 16);
  return a.v;
}
// B fragment from LDS: 32x16 bf16, lane l -> col; elems 0..7 = K[k0+8*hl ..], elems 8..15 = K[k0+16+8*hl ..]
__device__ __forceinline__ v16bf lds_b_frag(const unsigned short* lds, int colbase, int k0, int hl) {
  const unsigned short* p = lds + colbase + k0 + 8 * hl;
  Frag b;
  b.q[0] = *reinterpret_cast<const U128*>(p);
  b.q[1] = *reinterpret_cast<const U128*>(p + 16);
  return b.v;
}

// ---------------- prep kernels ----------------
__global__ void cvt_bf16_k(const float* __restrict__ in, unsigned short* __restrict__ out, size_t n) {
  size_t i = (size_t)blockIdx.x * blockDim.x + threadIdx.x;
  if (i < n) out[i] = f2bf(in[i]);
}

__global__ void build_A_k(const float* __restrict__ Wh, unsigned short* __restrict__ Am) {
  int k = blockIdx.x * blockDim.x + threadIdx.x;
  int n = blockIdx.y;
  if (k < H_DIM) {
    float v = Wh[(size_t)n * H_DIM + k] - Wh[(size_t)k * H_DIM + n];
    if (k == n) v -= GAMMA_C;
    Am[(size_t)n * H_DIM + k] = f2bf(v);
  }
}

__global__ void init_rec_k(const float* __restrict__ h0l, unsigned short* __restrict__ hbf,
                           unsigned* __restrict__ bar, int n) {
  int i = blockIdx.x * blockDim.x + threadIdx.x;
  if (i < n) hbf[i] = f2bf(h0l[i]);
  if (i == 0) *bar = 0u;
}

// ---------------- input-projection GEMM:  Out[m,n] = sum_k X[m,k]*W[n,k] + bias[n] ----------------
// grid: (M/64, N/64, 3 {z,r,h}); block 128 (4 waves); wave tile 16(M) x 64(N).
// The 64-column weight tile is staged once in LDS and shared by all 4 waves.
__global__ void __launch_bounds__(128) proj_gemm(
    const unsigned short* __restrict__ X,
    const unsigned short* __restrict__ Wz, const unsigned short* __restrict__ Wr, const unsigned short* __restrict__ Wv,
    const float* __restrict__ bz, const float* __restrict__ br, const float* __restrict__ bh,
    float* __restrict__ Pz, float* __restrict__ Pr, float* __restrict__ Ph,
    int M, int N, int K)
{
  extern __shared__ unsigned short smem[];           // 64 * (K+8) halfs
  const int sel = blockIdx.z;
  const unsigned short* W = (sel == 0) ? Wz : (sel == 1) ? Wr : Wv;
  const float* bias       = (sel == 0) ? bz : (sel == 1) ? br : bh;
  float* Out              = (sel == 0) ? Pz : (sel == 1) ? Pr : Ph;

  const int wave = threadIdx.x >> 5;
  const int lane = threadIdx.x & 31;
  const int hl   = lane >> 4;
  const int m0   = blockIdx.x * 64 + wave * 16;
  const int n0   = blockIdx.y * 64;
  const int arow = m0 + (lane & 15);
  const int PSTR = K + 8;
  const int KC   = K >> 3;                           // 16B chunks per column

  // cooperative stage of the 64-column weight tile into LDS (padded columns)
  for (int idx = threadIdx.x; idx < 64 * KC; idx += 128) {
    int c  = idx / KC;
    int kk = (idx - c * KC) << 3;
    *reinterpret_cast<U128*>(smem + c * PSTR + kk) =
        *reinterpret_cast<const U128*>(W + (size_t)(n0 + c) * K + kk);
  }
  __syncthreads();

  v8f acc[4];
#pragma unroll
  for (int j = 0; j < 4; ++j) acc[j] = zero8();

  for (int k0 = 0; k0 < K; k0 += 32) {
    v16bf a = load_a_frag(X, arow, K, k0, hl);
    if (k0 + 32 < K) __builtin_prefetch(X + (size_t)arow * K + k0 + 32, 0, 3);
#pragma unroll
    for (int j = 0; j < 4; ++j) {
      v16bf b = lds_b_frag(smem, (16 * j + (lane & 15)) * PSTR, k0, hl);
      acc[j] = __builtin_amdgcn_wmma_f32_16x16x32_bf16(false, a, false, b, (short)0, acc[j], false, false);
    }
  }
#pragma unroll
  for (int j = 0; j < 4; ++j) {
    int n = n0 + 16 * j + (lane & 15);
    float bv = bias[n];
#pragma unroll
    for (int v = 0; v < 8; ++v) {
      int m = m0 + v + 8 * hl;                       // C layout: VGPR v -> M = m0+v+8*hl
      Out[(size_t)m * N + n] = acc[j][v] + bv;
    }
  }
}

// ---------------- persistent recurrence kernel ----------------
__device__ __forceinline__ void grid_barrier(unsigned* bar, unsigned nwg, unsigned& phase) {
  __threadfence();
  __syncthreads();
  phase += nwg;
  if (threadIdx.x == 0) {
    atomicAdd(bar, 1u);
    volatile unsigned* vb = (volatile unsigned*)bar;
    while (*vb < phase) __builtin_amdgcn_s_sleep(1);
  }
  __syncthreads();
  __threadfence();
}

// grid: H/32 WGs (each owns a 32-column slice), block 128 (4 waves x 16 batch rows).
// The WG's Uz/Ur/A column slices (3 x 32 x 1024 bf16, ~193KB padded) are loaded
// into LDS ONCE via async-to-LDS DMA and reused for all 512 timesteps.
__global__ void __launch_bounds__(128) agru_rec(
    const float* __restrict__ Pz, const float* __restrict__ Pr, const float* __restrict__ Ph,
    const unsigned short* __restrict__ Uz, const unsigned short* __restrict__ Ur, const unsigned short* __restrict__ Am,
    const float* __restrict__ h0l, const float* __restrict__ epsp,
    unsigned short* __restrict__ hbf, unsigned short* __restrict__ rhbf,
    float* __restrict__ yout, unsigned short* __restrict__ ybf,
    float* __restrict__ hn_out,
    unsigned* __restrict__ bar, int nwg)
{
  extern __shared__ unsigned short smem[];           // 3 * 32 * RSTR halfs
  const float eps = *epsp;
  const int lane = threadIdx.x & 31;
  const int hl   = lane >> 4;
  const int m0   = (threadIdx.x >> 5) * 16;          // batch rows of this wave
  const int n0   = blockIdx.x * 32;                  // hidden-column slice
  const int arow = m0 + (lane & 15);

  // ---- async DMA: weights (Uz|Ur|A) slice -> LDS, 16B per lane per op ----
  {
    const int KC = H_DIM >> 3;                       // 128 chunks per column
    for (int idx = threadIdx.x; idx < 3 * 32 * KC; idx += 128) {
      int mat = idx / (32 * KC);
      int rem = idx - mat * (32 * KC);
      int c   = rem / KC;
      int kk  = (rem - c * KC) << 3;
      const unsigned short* src =
          ((mat == 0) ? Uz : (mat == 1) ? Ur : Am) + (size_t)(n0 + c) * H_DIM + kk;
      unsigned dst = (unsigned)(((mat * 32 + c) * RSTR + kk) * 2);
      unsigned long long ga = (unsigned long long)src;
      asm volatile("global_load_async_to_lds_b128 %0, %1, off"
                   :: "v"(dst), "v"(ga) : "memory");
    }
    asm volatile("s_wait_asynccnt 0" ::: "memory");
    __syncthreads();
  }

  float hreg[2][8];                                  // fp32 h register-resident all 512 steps
  float zreg[2][8];
#pragma unroll
  for (int j = 0; j < 2; ++j) {
    int n = n0 + 16 * j + (lane & 15);
#pragma unroll
    for (int v = 0; v < 8; ++v)
      hreg[j][v] = h0l[(size_t)(m0 + v + 8 * hl) * H_DIM + n];
  }

  unsigned phase = 0;
  for (int t = 0; t < S_LEN; ++t) {
    // ---- stage 1: z = sig(xz + h Uz^T), r = sig(xr + h Ur^T), publish r*h (bf16) ----
    v8f az[2], ar[2];
    az[0] = zero8(); az[1] = zero8(); ar[0] = zero8(); ar[1] = zero8();
    for (int k0 = 0; k0 < H_DIM; k0 += 32) {
      v16bf a = load_a_frag(hbf, arow, H_DIM, k0, hl);
#pragma unroll
      for (int j = 0; j < 2; ++j) {
        int cb = (16 * j + (lane & 15)) * RSTR;
        v16bf bzf = lds_b_frag(smem, cb, k0, hl);                    // Uz slice
        az[j] = __builtin_amdgcn_wmma_f32_16x16x32_bf16(false, a, false, bzf, (short)0, az[j], false, false);
        v16bf brf = lds_b_frag(smem + 32 * RSTR, cb, k0, hl);        // Ur slice
        ar[j] = __builtin_amdgcn_wmma_f32_16x16x32_bf16(false, a, false, brf, (short)0, ar[j], false, false);
      }
    }
#pragma unroll
    for (int j = 0; j < 2; ++j) {
      int n = n0 + 16 * j + (lane & 15);
#pragma unroll
      for (int v = 0; v < 8; ++v) {
        int b = m0 + v + 8 * hl;
        size_t pidx = ((size_t)t * B_DIM + b) * H_DIM + n;
        float zv = sigm(Pz[pidx] + az[j][v]);
        float rv = sigm(Pr[pidx] + ar[j][v]);
        zreg[j][v] = zv;
        rhbf[(size_t)b * H_DIM + n] = f2bf(rv * hreg[j][v]);
      }
    }
    grid_barrier(bar, (unsigned)nwg, phase);

    // ---- stage 2: dh = tanh(xh + (r*h) A^T), h += eps*z*dh ----
    v8f ah[2];
    ah[0] = zero8(); ah[1] = zero8();
    for (int k0 = 0; k0 < H_DIM; k0 += 32) {
      v16bf a = load_a_frag(rhbf, arow, H_DIM, k0, hl);
#pragma unroll
      for (int j = 0; j < 2; ++j) {
        int cb = (16 * j + (lane & 15)) * RSTR;
        v16bf bf = lds_b_frag(smem + 64 * RSTR, cb, k0, hl);         // A slice
        ah[j] = __builtin_amdgcn_wmma_f32_16x16x32_bf16(false, a, false, bf, (short)0, ah[j], false, false);
      }
    }
#pragma unroll
    for (int j = 0; j < 2; ++j) {
      int n = n0 + 16 * j + (lane & 15);
#pragma unroll
      for (int v = 0; v < 8; ++v) {
        int b = m0 + v + 8 * hl;
        size_t pidx = ((size_t)t * B_DIM + b) * H_DIM + n;
        float dh = tanhf(Ph[pidx] + ah[j][v]);
        float hn = hreg[j][v] + eps * zreg[j][v] * dh;
        hreg[j][v] = hn;
        hbf[(size_t)b * H_DIM + n] = f2bf(hn);
        if (yout) yout[pidx] = hn;
        if (ybf)  ybf[pidx]  = f2bf(hn);
      }
    }
    grid_barrier(bar, (unsigned)nwg, phase);
  }
#pragma unroll
  for (int j = 0; j < 2; ++j) {
    int n = n0 + 16 * j + (lane & 15);
#pragma unroll
    for (int v = 0; v < 8; ++v)
      hn_out[(size_t)(m0 + v + 8 * hl) * H_DIM + n] = hreg[j][v];
  }
}

// ---------------- host launcher ----------------
extern "C" void kernel_launch(void* const* d_in, const int* in_sizes, int n_in,
                              void* d_out, int out_size, void* d_ws, size_t ws_size,
                              hipStream_t stream) {
  const float* x    = (const float*)d_in[0];
  const float* h0   = (const float*)d_in[1];
  const float* Wz0  = (const float*)d_in[2];
  const float* bz0  = (const float*)d_in[3];
  const float* Uz0  = (const float*)d_in[4];
  const float* Wr0  = (const float*)d_in[5];
  const float* br0  = (const float*)d_in[6];
  const float* Ur0  = (const float*)d_in[7];
  const float* Vh0  = (const float*)d_in[8];
  const float* bh0  = (const float*)d_in[9];
  const float* Wh0  = (const float*)d_in[10];
  const float* eps0 = (const float*)d_in[11];
  const float* Wz1  = (const float*)d_in[12];
  const float* bz1  = (const float*)d_in[13];
  const float* Uz1  = (const float*)d_in[14];
  const float* Wr1  = (const float*)d_in[15];
  const float* br1  = (const float*)d_in[16];
  const float* Ur1  = (const float*)d_in[17];
  const float* Vh1  = (const float*)d_in[18];
  const float* bh1  = (const float*)d_in[19];
  const float* Wh1  = (const float*)d_in[20];
  const float* eps1 = (const float*)d_in[21];

  const size_t S = S_LEN, B = B_DIM, Iw = I_DIM, H = H_DIM;
  const size_t M = S * B;

  char* base = (char*)d_ws;
  size_t off = 0;
  auto carve = [&](size_t bytes) -> char* {
    char* p = base + off;
    off = (off + bytes + 255) & ~(size_t)255;
    return p;
  };
  unsigned short* xbf  = (unsigned short*)carve(M * Iw * 2);
  unsigned short* y0bf = (unsigned short*)carve(M * H * 2);
  float* Pz = (float*)carve(M * H * 4);
  float* Pr = (float*)carve(M * H * 4);
  float* Ph = (float*)carve(M * H * 4);
  unsigned short* wz0b = (unsigned short*)carve(H * Iw * 2);
  unsigned short* wr0b = (unsigned short*)carve(H * Iw * 2);
  unsigned short* vh0b = (unsigned short*)carve(H * Iw * 2);
  unsigned short* uz0b = (unsigned short*)carve(H * H * 2);
  unsigned short* ur0b = (unsigned short*)carve(H * H * 2);
  unsigned short* a0b  = (unsigned short*)carve(H * H * 2);
  unsigned short* wz1b = (unsigned short*)carve(H * H * 2);
  unsigned short* wr1b = (unsigned short*)carve(H * H * 2);
  unsigned short* vh1b = (unsigned short*)carve(H * H * 2);
  unsigned short* uz1b = (unsigned short*)carve(H * H * 2);
  unsigned short* ur1b = (unsigned short*)carve(H * H * 2);
  unsigned short* a1b  = (unsigned short*)carve(H * H * 2);
  unsigned short* hbf  = (unsigned short*)carve(B * H * 2);
  unsigned short* rhbf = (unsigned short*)carve(B * H * 2);
  unsigned* bar        = (unsigned*)carve(256);

  float* y1  = (float*)d_out;               // (S,B,H)
  float* hn0 = (float*)d_out + M * H;       // h_n layer 0
  float* hn1 = hn0 + B * H;                 // h_n layer 1

  auto cvt = [&](const float* in, unsigned short* out, size_t n) {
    int blocks = (int)((n + 255) / 256);
    hipLaunchKernelGGL(cvt_bf16_k, dim3(blocks), dim3(256), 0, stream, in, out, n);
  };
  cvt(Wz0, wz0b, H * Iw); cvt(Wr0, wr0b, H * Iw); cvt(Vh0, vh0b, H * Iw);
  cvt(Uz0, uz0b, H * H);  cvt(Ur0, ur0b, H * H);
  cvt(Wz1, wz1b, H * H);  cvt(Wr1, wr1b, H * H);  cvt(Vh1, vh1b, H * H);
  cvt(Uz1, uz1b, H * H);  cvt(Ur1, ur1b, H * H);
  cvt(x, xbf, M * Iw);
  hipLaunchKernelGGL(build_A_k, dim3((unsigned)(H / 256), (unsigned)H), dim3(256), 0, stream, Wh0, a0b);
  hipLaunchKernelGGL(build_A_k, dim3((unsigned)(H / 256), (unsigned)H), dim3(256), 0, stream, Wh1, a1b);

  const int NWG = (int)(H / 32);
  const size_t recLds   = 3u * 32u * RSTR * 2u;          // ~193.5 KB
  const size_t projLds0 = 64u * (Iw + 8) * 2u;           // 65 KB (K=512)
  const size_t projLds1 = 64u * (H + 8) * 2u;            // 129 KB (K=1024)

  // ---- layer 0 ----
  hipLaunchKernelGGL(proj_gemm, dim3((unsigned)(M / 64), (unsigned)(H / 64), 3), dim3(128), projLds0, stream,
                     xbf, wz0b, wr0b, vh0b, bz0, br0, bh0, Pz, Pr, Ph, (int)M, (int)H, (int)Iw);
  hipLaunchKernelGGL(init_rec_k, dim3((unsigned)((B * H + 255) / 256)), dim3(256), 0, stream,
                     h0, hbf, bar, (int)(B * H));
  hipLaunchKernelGGL(agru_rec, dim3((unsigned)NWG), dim3(128), recLds, stream,
                     Pz, Pr, Ph, uz0b, ur0b, a0b, h0, eps0,
                     hbf, rhbf, (float*)nullptr, y0bf, hn0, bar, NWG);

  // ---- layer 1 ----
  hipLaunchKernelGGL(proj_gemm, dim3((unsigned)(M / 64), (unsigned)(H / 64), 3), dim3(128), projLds1, stream,
                     y0bf, wz1b, wr1b, vh1b, bz1, br1, bh1, Pz, Pr, Ph, (int)M, (int)H, (int)H);
  hipLaunchKernelGGL(init_rec_k, dim3((unsigned)((B * H + 255) / 256)), dim3(256), 0, stream,
                     h0 + B * H, hbf, bar, (int)(B * H));
  hipLaunchKernelGGL(agru_rec, dim3((unsigned)NWG), dim3(128), recLds, stream,
                     Pz, Pr, Ph, uz1b, ur1b, a1b, h0 + B * H, eps1,
                     hbf, rhbf, y1, (unsigned short*)nullptr, hn1, bar, NWG);

  (void)in_sizes; (void)n_in; (void)out_size; (void)ws_size;
}
